// MultiHeadAttention_55336358642102
// MI455X (gfx1250) — matrix operand order repaired and checked by reading, not compiled, not run on hardware
//
#include <hip/hip_runtime.h>
#include <math.h>

// ---------------------------------------------------------------------------
// Types for CDNA5 WMMA
// ---------------------------------------------------------------------------
typedef __bf16 v16bf __attribute__((ext_vector_type(16)));
typedef float  v8f   __attribute__((ext_vector_type(8)));

__device__ __forceinline__ v8f wmma_bf16(v16bf a, v16bf b, v8f c) {
    // D = A(16x32 bf16) * B(32x16 bf16) + C(16x16 f32)
    return __builtin_amdgcn_wmma_f32_16x16x32_bf16(
        false, a, false, b, (short)0, c, false, false);
}

__device__ __forceinline__ v8f zero8() {
    v8f z;
    for (int i = 0; i < 8; ++i) z[i] = 0.0f;
    return z;
}

__device__ __forceinline__ __bf16 f2bf(float f) {
    union { float f; unsigned int u; } a; a.f = f;
    unsigned int u = a.u;
    u += 0x7FFFu + ((u >> 16) & 1u);          // round-to-nearest-even
    unsigned short s = (unsigned short)(u >> 16);
    union { unsigned short s; __bf16 b; } r; r.s = s;
    return r.b;
}

// CDNA5 async global->LDS copy (16 bytes per lane), tracked by ASYNCcnt.
__device__ __forceinline__ void async_ld16(unsigned lds_off, const void* gaddr) {
    asm volatile("global_load_async_to_lds_b128 %0, %1, off"
                 :: "v"(lds_off), "v"(gaddr) : "memory");
}
__device__ __forceinline__ void wait_async0() {
    asm volatile("s_wait_asynccnt 0x0" ::: "memory");
}
__device__ __forceinline__ void wait_async4() {
    // allow the 4 just-issued prefetch loads of the *next* tile to stay in
    // flight; everything older (current tile) is complete.
    asm volatile("s_wait_asynccnt 0x4" ::: "memory");
}

// Problem constants
#define HN 8
#define DMODEL 1024
#define DKV 128
#define DPOS 256
#define DCONT 768
#define BB 8
#define LL 1024
#define TT (BB * LL)

// ---------------------------------------------------------------------------
// K0: convert input f32 -> bf16 (row-major T x 1024)
// ---------------------------------------------------------------------------
__global__ void conv_x_kernel(const float* __restrict__ x, __bf16* __restrict__ xb) {
    size_t i = (size_t)blockIdx.x * blockDim.x + threadIdx.x;
    if (i < (size_t)TT * DMODEL) xb[i] = f2bf(x[i]);
}

// ---------------------------------------------------------------------------
// K_wprep: transposed, zero-padded QKV weight: Wt[n][k], n in [0,3072),
// k in [0,1024). n = type*1024 + h*128 + d.  d<64 -> content rows (k<768),
// d>=64 -> position rows (k>=768).
// ---------------------------------------------------------------------------
__global__ void prep_wqkv_kernel(const float* __restrict__ qs1, const float* __restrict__ ks1,
                                 const float* __restrict__ vs1, const float* __restrict__ qs2,
                                 const float* __restrict__ ks2, const float* __restrict__ vs2,
                                 __bf16* __restrict__ wt) {
    size_t idx = (size_t)blockIdx.x * blockDim.x + threadIdx.x;
    if (idx >= (size_t)3072 * 1024) return;
    int kcol = (int)(idx & 1023);
    int n    = (int)(idx >> 10);
    int type = n >> 10;
    int r    = n & 1023;
    int h    = r >> 7;
    int d    = r & 127;
    const float* w1 = (type == 0) ? qs1 : (type == 1) ? ks1 : vs1;
    const float* w2 = (type == 0) ? qs2 : (type == 1) ? ks2 : vs2;
    float v = 0.0f;
    if (d < 64) {
        if (kcol < DCONT) v = w1[((size_t)h * DCONT + kcol) * 64 + d];
    } else {
        if (kcol >= DCONT) v = w2[((size_t)h * DPOS + (kcol - DCONT)) * 64 + (d - 64)];
    }
    wt[idx] = f2bf(v);
}

// ---------------------------------------------------------------------------
// K_pprep: output-projection weight, transposed + zero-padded: Wt[n][k].
// ---------------------------------------------------------------------------
__global__ void prep_wp_kernel(const float* __restrict__ p1, const float* __restrict__ p2,
                               __bf16* __restrict__ wt) {
    size_t idx = (size_t)blockIdx.x * blockDim.x + threadIdx.x;
    if (idx >= (size_t)1024 * 1024) return;
    int kcol = (int)(idx & 1023);
    int n    = (int)(idx >> 10);
    int h    = kcol >> 7;
    int d    = kcol & 127;
    float v = 0.0f;
    if (n < DCONT) {
        if (d < 64) v = p1[((size_t)(h * 64 + d)) * DCONT + n];
    } else {
        if (d >= 64) v = p2[((size_t)(h * 64 + (d - 64))) * DPOS + (n - DCONT)];
    }
    wt[idx] = f2bf(v);
}

// ---------------------------------------------------------------------------
// Shared GEMM core: 128x128 block tile, 8 waves (wave tile 32x64), K=1024,
// double-buffered async global->LDS staging (ASYNCcnt pipelining).
// acc layout: acc[i][jn] = 16x16 f32 tile at (wr*32+i*16, wc*64+jn*16).
// ---------------------------------------------------------------------------
#define GEMM_CORE(Aptr, Bptr)                                                          \
    const int K = 1024;                                                                \
    __shared__ __attribute__((aligned(16))) __bf16 As[2][128][40];                     \
    __shared__ __attribute__((aligned(16))) __bf16 Bs[2][128][40];                     \
    int tid  = threadIdx.x;                                                            \
    int wave = tid >> 5, lane = tid & 31;                                              \
    int l = lane & 15, hh = lane >> 4;                                                 \
    int wr = wave & 3, wc = wave >> 2;                                                 \
    size_t mbase = (size_t)blockIdx.x * 128;                                           \
    size_t nbase = (size_t)blockIdx.y * 128;                                           \
    int row0 = tid >> 2,         co0 = (tid & 3) * 8;                                  \
    int row1 = (tid + 256) >> 2, co1 = (tid & 3) * 8;                                  \
    v8f acc[2][4];                                                                     \
    _Pragma("unroll") for (int i = 0; i < 2; ++i)                                      \
        _Pragma("unroll") for (int jn = 0; jn < 4; ++jn) acc[i][jn] = zero8();         \
    /* prologue: stage k-tile 0 into buffer 0 */                                       \
    async_ld16((unsigned)(size_t)&As[0][row0][co0], (const void*)(Aptr + (mbase + row0) * K + co0)); \
    async_ld16((unsigned)(size_t)&As[0][row1][co1], (const void*)(Aptr + (mbase + row1) * K + co1)); \
    async_ld16((unsigned)(size_t)&Bs[0][row0][co0], (const void*)(Bptr + (nbase + row0) * K + co0)); \
    async_ld16((unsigned)(size_t)&Bs[0][row1][co1], (const void*)(Bptr + (nbase + row1) * K + co1)); \
    int cur = 0;                                                                       \
    for (int k0 = 0; k0 < K; k0 += 32) {                                               \
        int nxt = k0 + 32;                                                             \
        if (nxt < K) { /* prefetch next tile into the other buffer */                  \
            int nb = cur ^ 1;                                                          \
            async_ld16((unsigned)(size_t)&As[nb][row0][co0], (const void*)(Aptr + (mbase + row0) * K + nxt + co0)); \
            async_ld16((unsigned)(size_t)&As[nb][row1][co1], (const void*)(Aptr + (mbase + row1) * K + nxt + co1)); \
            async_ld16((unsigned)(size_t)&Bs[nb][row0][co0], (const void*)(Bptr + (nbase + row0) * K + nxt + co0)); \
            async_ld16((unsigned)(size_t)&Bs[nb][row1][co1], (const void*)(Bptr + (nbase + row1) * K + nxt + co1)); \
            wait_async4();  /* current tile resident, prefetch in flight */            \
        } else {                                                                       \
            wait_async0();                                                             \
        }                                                                              \
        __syncthreads();   /* current tile visible to all waves */                     \
        _Pragma("unroll") for (int i = 0; i < 2; ++i) {                                \
            v16bf a;                                                                   \
            int mrow = wr * 32 + i * 16 + l;                                           \
            _Pragma("unroll") for (int j = 0; j < 8; ++j) {                            \
                int kb = 2 * j + 16 * hh;                          /* A-layout */      \
                a[2 * j]     = As[cur][mrow][kb];                                      \
                a[2 * j + 1] = As[cur][mrow][kb + 1];                                  \
            }                                                                          \
            _Pragma("unroll") for (int jn = 0; jn < 4; ++jn) {                         \
                v16bf b;                                                               \
                int nrow = wc * 64 + jn * 16 + l;                                      \
                _Pragma("unroll") for (int j = 0; j < 8; ++j) {                        \
                    int kb = 2 * j + 16 * hh;                      /* B-layout */      \
                    b[2 * j]     = Bs[cur][nrow][kb];                                  \
                    b[2 * j + 1] = Bs[cur][nrow][kb + 1];                              \
                }                                                                      \
                acc[i][jn] = wmma_bf16(a, b, acc[i][jn]);                              \
            }                                                                          \
        }                                                                              \
        __syncthreads();   /* all waves done reading before buffer reuse */            \
        cur ^= 1;                                                                      \
    }

// ---------------------------------------------------------------------------
// K1: QKV GEMM.  X(8192 x 1024) bf16  @  Wt^T (K=1024, N=3072) -> Q,K,Vt bf16
// Epilogue scatters: Q,K row-major (H,T,128); V transposed (H,B,128,L).
// ---------------------------------------------------------------------------
__global__ __launch_bounds__(256) void gemm_qkv_kernel(
        const __bf16* __restrict__ X, const __bf16* __restrict__ Wt,
        __bf16* __restrict__ q, __bf16* __restrict__ kk, __bf16* __restrict__ vt) {
    GEMM_CORE(X, Wt)

    // Output type (q/k/v) is uniform per block (N tile of 128 lies within one
    // 1024-column group) -> branch once, store branch-free.
    int type = (int)(nbase >> 10);
    size_t nloc = nbase & 1023;
    if (type == 0) {
#pragma unroll
        for (int i = 0; i < 2; ++i)
#pragma unroll
            for (int jn = 0; jn < 4; ++jn)
#pragma unroll
                for (int j = 0; j < 8; ++j) {
                    size_t m = mbase + wr * 32 + i * 16 + j + 8 * hh;
                    size_t n = nloc + wc * 64 + jn * 16 + l;
                    q[(((n >> 7) * TT + m) << 7) + (n & 127)] = f2bf(acc[i][jn][j]);
                }
    } else if (type == 1) {
#pragma unroll
        for (int i = 0; i < 2; ++i)
#pragma unroll
            for (int jn = 0; jn < 4; ++jn)
#pragma unroll
                for (int j = 0; j < 8; ++j) {
                    size_t m = mbase + wr * 32 + i * 16 + j + 8 * hh;
                    size_t n = nloc + wc * 64 + jn * 16 + l;
                    kk[(((n >> 7) * TT + m) << 7) + (n & 127)] = f2bf(acc[i][jn][j]);
                }
    } else {
#pragma unroll
        for (int i = 0; i < 2; ++i)
#pragma unroll
            for (int jn = 0; jn < 4; ++jn)
#pragma unroll
                for (int j = 0; j < 8; ++j) {
                    size_t m = mbase + wr * 32 + i * 16 + j + 8 * hh;
                    size_t n = nloc + wc * 64 + jn * 16 + l;
                    size_t hd = n >> 7, d = n & 127;
                    size_t b = m >> 10, pos = m & 1023;
                    vt[(((hd * BB + b) << 7) + d) * LL + pos] = f2bf(acc[i][jn][j]);
                }
    }
}

// ---------------------------------------------------------------------------
// K2: flash attention per (h,b). 1024 blocks x 128 threads (4 waves).
// Wave handles 16 query rows; streams 32 keys/iter with online softmax.
// ---------------------------------------------------------------------------
__global__ __launch_bounds__(128) void attn_kernel(
        const __bf16* __restrict__ q, const __bf16* __restrict__ k,
        const __bf16* __restrict__ vt, __bf16* __restrict__ att) {
    int wave = threadIdx.x >> 5;
    int lane = threadIdx.x & 31;
    int l = lane & 15, hh = lane >> 4;
    int blk = blockIdx.x;
    int mt = blk & 15;
    int hb = blk >> 4;
    int b = hb & 7, h = hb >> 3;
    int qrow0 = mt * 64 + wave * 16;

    const __bf16* qbase = q  + ((size_t)h * TT + (size_t)b * LL) * DKV;
    const __bf16* kbase = k  + ((size_t)h * TT + (size_t)b * LL) * DKV;
    const __bf16* vbase = vt + ((size_t)(h * BB + b)) * DKV * LL;

    __shared__ float  Sst[4][16][32];
    __shared__ __bf16 Pst[4][16][32];
    __shared__ float  mrow[4][16], lrow[4][16], srow[4][16];

    // Q fragments: 16 rows x 128 d, four 32-d A-fragments
    v16bf qf[4];
#pragma unroll
    for (int c = 0; c < 4; ++c)
#pragma unroll
        for (int j = 0; j < 8; ++j) {
            int kb = 2 * j + 16 * hh;
            const __bf16* p = qbase + ((size_t)(qrow0 + l)) * DKV + c * 32 + kb;
            qf[c][2 * j] = p[0]; qf[c][2 * j + 1] = p[1];
        }

    v8f oacc[8];
#pragma unroll
    for (int ns = 0; ns < 8; ++ns) oacc[ns] = zero8();

    if (lane < 16) { mrow[wave][lane] = -1e30f; lrow[wave][lane] = 0.0f; }
    __syncthreads();

    const float inv_temper = 1.0f / 32.0f;   // 1/sqrt(1024)

    for (int kt = 0; kt < LL; kt += 32) {
        // ---- S = Q * K^T for 32 keys (two 16-col WMMA tiles) ----
        v8f s[2]; s[0] = zero8(); s[1] = zero8();
#pragma unroll
        for (int nt = 0; nt < 2; ++nt)
#pragma unroll
            for (int c = 0; c < 4; ++c) {
                v16bf bf;
#pragma unroll
                for (int j = 0; j < 8; ++j) {
                    int kb = 2 * j + 16 * hh;                     // pair along d
                    const __bf16* p = kbase + ((size_t)(kt + nt * 16 + l)) * DKV + c * 32 + kb;
                    bf[2 * j] = p[0]; bf[2 * j + 1] = p[1];
                }
                s[nt] = wmma_bf16(qf[c], bf, s[nt]);
            }
#pragma unroll
        for (int nt = 0; nt < 2; ++nt)
#pragma unroll
            for (int j = 0; j < 8; ++j)
                Sst[wave][j + 8 * hh][nt * 16 + l] = s[nt][j] * inv_temper;
        __syncthreads();

        // ---- online softmax: all 32 lanes (row = l, column half = hh) ----
        {
            float mold = mrow[wave][l];
            float cmax = -1e30f;
#pragma unroll
            for (int c2 = 0; c2 < 16; ++c2)
                cmax = fmaxf(cmax, Sst[wave][l][hh * 16 + c2]);
            cmax = fmaxf(cmax, __shfl_xor(cmax, 16, 32));
            float mnew = fmaxf(mold, cmax);
            float sc = __expf(mold - mnew);
            float psum = 0.0f;
#pragma unroll
            for (int c2 = 0; c2 < 16; ++c2) {
                float p = __expf(Sst[wave][l][hh * 16 + c2] - mnew);
                psum += p;
                Pst[wave][l][hh * 16 + c2] = f2bf(p);
            }
            psum += __shfl_xor(psum, 16, 32);
            if (lane < 16) {
                mrow[wave][l] = mnew;
                lrow[wave][l] = lrow[wave][l] * sc + psum;
                srow[wave][l] = sc;
            }
        }
        __syncthreads();

        // ---- rescale O, then O += P * V ----
        float scv[8];
#pragma unroll
        for (int j = 0; j < 8; ++j) scv[j] = srow[wave][j + 8 * hh];
#pragma unroll
        for (int ns = 0; ns < 8; ++ns)
#pragma unroll
            for (int j = 0; j < 8; ++j) oacc[ns][j] *= scv[j];

        v16bf pf;
#pragma unroll
        for (int j = 0; j < 8; ++j) {
            int kb = 2 * j + 16 * hh;                              // pair along keys
            pf[2 * j]     = Pst[wave][l][kb];
            pf[2 * j + 1] = Pst[wave][l][kb + 1];
        }
#pragma unroll
        for (int ns = 0; ns < 8; ++ns) {
            v16bf vf;
#pragma unroll
            for (int j = 0; j < 8; ++j) {
                int kb = 2 * j + 16 * hh;                          // pair along keys
                const __bf16* p = vbase + ((size_t)(ns * 16 + l)) * LL + kt + kb;
                vf[2 * j] = p[0]; vf[2 * j + 1] = p[1];
            }
            oacc[ns] = wmma_bf16(pf, vf, oacc[ns]);
        }
        __syncthreads();
    }

    // ---- finalize: divide by row sum, write att (T x H*128) ----
    float inv[8];
#pragma unroll
    for (int j = 0; j < 8; ++j) inv[j] = 1.0f / lrow[wave][j + 8 * hh];
#pragma unroll
    for (int ns = 0; ns < 8; ++ns)
#pragma unroll
        for (int j = 0; j < 8; ++j) {
            size_t t = (size_t)b * LL + qrow0 + j + 8 * hh;
            size_t col = (size_t)h * DKV + ns * 16 + l;
            att[t * DMODEL + col] = f2bf(oacc[ns][j] * inv[j]);
        }
}

// ---------------------------------------------------------------------------
// K3: output projection GEMM att(8192x1024) @ Wp (K=1024,N=1024) + residual.
// Writes f32 z = proj + inp into d_out (LayerNorm applied in-place by K4).
// ---------------------------------------------------------------------------
__global__ __launch_bounds__(256) void gemm_proj_kernel(
        const __bf16* __restrict__ A, const __bf16* __restrict__ Wt,
        const float* __restrict__ resid, float* __restrict__ out) {
    GEMM_CORE(A, Wt)

#pragma unroll
    for (int i = 0; i < 2; ++i)
#pragma unroll
        for (int jn = 0; jn < 4; ++jn)
#pragma unroll
            for (int j = 0; j < 8; ++j) {
                size_t m = mbase + wr * 32 + i * 16 + j + 8 * hh;
                size_t n = nbase + wc * 64 + jn * 16 + l;
                out[m * DMODEL + n] = acc[i][jn][j] + resid[m * DMODEL + n];
            }
}

// ---------------------------------------------------------------------------
// K4: in-place LayerNorm on d_out: (z-mu)/(sigma_ddof1 + eps) * a + b
// ---------------------------------------------------------------------------
__global__ __launch_bounds__(256) void layernorm_kernel(
        float* __restrict__ out, const float* __restrict__ ga, const float* __restrict__ be) {
    __shared__ float red[256], red2[256];
    int row = blockIdx.x;
    int tid = threadIdx.x;
    float* z = out + (size_t)row * DMODEL;
    float v[4], s = 0.0f, s2 = 0.0f;
#pragma unroll
    for (int i = 0; i < 4; ++i) {
        v[i] = z[tid + i * 256];
        s  += v[i];
        s2 += v[i] * v[i];
    }
    red[tid] = s; red2[tid] = s2;
    __syncthreads();
    for (int o = 128; o > 0; o >>= 1) {
        if (tid < o) { red[tid] += red[tid + o]; red2[tid] += red2[tid + o]; }
        __syncthreads();
    }
    float mean = red[0] * (1.0f / 1024.0f);
    float var  = fmaxf((red2[0] - 1024.0f * mean * mean) * (1.0f / 1023.0f), 0.0f);
    float denom = 1.0f / (sqrtf(var) + 0.001f);
#pragma unroll
    for (int i = 0; i < 4; ++i) {
        int c = tid + i * 256;
        z[c] = (v[i] - mean) * denom * ga[c] + be[c];
    }
}

// ---------------------------------------------------------------------------
// Launcher
// ---------------------------------------------------------------------------
extern "C" void kernel_launch(void* const* d_in, const int* in_sizes, int n_in,
                              void* d_out, int out_size, void* d_ws, size_t ws_size,
                              hipStream_t stream) {
    const float* inp  = (const float*)d_in[0];
    const float* qs1  = (const float*)d_in[1];
    const float* ks1  = (const float*)d_in[2];
    const float* vs1  = (const float*)d_in[3];
    const float* qs2  = (const float*)d_in[4];
    const float* ks2  = (const float*)d_in[5];
    const float* vs2  = (const float*)d_in[6];
    const float* wp1  = (const float*)d_in[7];
    const float* wp2  = (const float*)d_in[8];
    const float* ln_a = (const float*)d_in[9];
    const float* ln_b = (const float*)d_in[10];
    (void)in_sizes; (void)n_in; (void)out_size; (void)ws_size;

    // Workspace layout (72 MB total):
    //   [0,16MB)   X bf16 (later reused as attention output)
    //   [16,32MB)  Q bf16 (H,T,128)
    //   [32,48MB)  K bf16 (H,T,128)
    //   [48,64MB)  V^T bf16 (H,B,128,L)
    //   [64,70MB)  Wqkv^T bf16 (3072 x 1024)
    //   [70,72MB)  Wp^T  bf16 (1024 x 1024)
    char* ws = (char*)d_ws;
    __bf16* xatt = (__bf16*)(ws);
    __bf16* qb   = (__bf16*)(ws + ((size_t)16 << 20));
    __bf16* kb   = (__bf16*)(ws + ((size_t)32 << 20));
    __bf16* vtb  = (__bf16*)(ws + ((size_t)48 << 20));
    __bf16* wqkv = (__bf16*)(ws + ((size_t)64 << 20));
    __bf16* wp   = (__bf16*)(ws + ((size_t)70 << 20));
    float*  out  = (float*)d_out;

    conv_x_kernel<<<(TT * DMODEL) / 256, 256, 0, stream>>>(inp, xatt);
    prep_wqkv_kernel<<<(3072 * 1024) / 256, 256, 0, stream>>>(qs1, ks1, vs1, qs2, ks2, vs2, wqkv);
    prep_wp_kernel<<<(1024 * 1024) / 256, 256, 0, stream>>>(wp1, wp2, wp);

    gemm_qkv_kernel<<<dim3(TT / 128, 3072 / 128), 256, 0, stream>>>(xatt, wqkv, qb, kb, vtb);
    attn_kernel<<<HN * BB * (LL / 64), 128, 0, stream>>>(qb, kb, vtb, xatt);
    gemm_proj_kernel<<<dim3(TT / 128, DMODEL / 128), 256, 0, stream>>>(xatt, wp, inp, out);
    layernorm_kernel<<<TT, 256, 0, stream>>>(out, ln_a, ln_b);
}
